// ScaledDotProductAttention_23227183137497
// MI455X (gfx1250) — compile-verified, matrix operand order checked by reading, not run on hardware
//
#include <hip/hip_runtime.h>
#include <cstdint>

// MI455X / gfx1250, wave32. WMMA bf16 hi/lo-split (3-product) fp32 emulation.

typedef __attribute__((ext_vector_type(16))) __bf16 v16bf;
typedef __attribute__((ext_vector_type(8)))  float  v8f;

#define B_       8
#define LQ_      2048
#define LK_      2048
#define D_       64
#define INV_TEMP 0.125f
#define NEG_BIG  (-3.0e38f)   // finite "-inf": exp underflows to 0, no NaN paths

// D = A*B + C with A,B split into bf16 hi/lo; hi*hi + hi*lo + lo*hi ~ fp32.
static __device__ __forceinline__ v8f wmma_bf16x3(v8f c, v16bf ah, v16bf al,
                                                  v16bf bh, v16bf bl) {
  c = __builtin_amdgcn_wmma_f32_16x16x32_bf16(false, ah, false, bh, (short)0, c, false, false);
  c = __builtin_amdgcn_wmma_f32_16x16x32_bf16(false, ah, false, bl, (short)0, c, false, false);
  c = __builtin_amdgcn_wmma_f32_16x16x32_bf16(false, al, false, bh, (short)0, c, false, false);
  return c;
}

// Load one A/B fragment (16x32 / 32x16 bf16) with hi/lo split from fp32 memory.
// ISA 7.12.2 layout: lane holds M(or N)=lane%16; contraction index set is
// {0..7,16..23} for lanes 0-15 and {8..15,24..31} for lanes 16-31.
// kk(j) = (j>>3)*16 + kbase + (j&7),  kbase = (lane>=16) ? 8 : 0.
static __device__ __forceinline__ void split_frag(const float* __restrict__ p,
                                                  int kbase, int stride,
                                                  v16bf& hi, v16bf& lo) {
#pragma unroll
  for (int j = 0; j < 16; ++j) {
    int kk = ((j >> 3) << 4) + kbase + (j & 7);
    float x = p[(size_t)kk * stride];
    __bf16 h = (__bf16)x;
    hi[j] = h;
    lo[j] = (__bf16)(x - (float)h);
  }
}

// ---------------------------------------------------------------------------
// Kernel 1: S = (Q@K^T + f1..f5)/T, masked -> raw scores into attn region,
//           plus per-row online-softmax (max, sum) stats into d_ws.
// Grid: (LQ/16, B). Block: 256 (8 waves). Wave w owns k-col blocks w, w+8, ...
// ---------------------------------------------------------------------------
__global__ __launch_bounds__(256) void qk_scores_kernel(
    const float* __restrict__ q, const float* __restrict__ k,
    const uint8_t* __restrict__ mask,
    const float* __restrict__ f1, const float* __restrict__ f2,
    const float* __restrict__ f3, const float* __restrict__ f4,
    const float* __restrict__ f5,
    float* __restrict__ attn, float* __restrict__ stats) {
  __shared__ float red_m[8][16];
  __shared__ float red_s[8][16];

  const int b    = blockIdx.y;
  const int q0   = blockIdx.x * 16;
  const int tid  = threadIdx.x;
  const int wave = tid >> 5;
  const int lane = tid & 31;
  const int lm   = lane & 15;     // A: row M ; B: col N ; C: col N
  const int half = lane >> 4;     // 0 | 1
  const int kbase = half * 8;     // contraction-index base

  const size_t bq0 = (size_t)b * LQ_ + q0;

  // Q fragments: two 32-wide K chunks cover D=64; loaded once per block.
  const float* qrow = q + (bq0 + lm) * D_;
  v16bf qh[2], ql[2];
  split_frag(qrow + 0,  kbase, 1, qh[0], ql[0]);
  split_frag(qrow + 32, kbase, 1, qh[1], ql[1]);

  // Online softmax stats; per-lane slot i covers row (i + half*8), this
  // lane's columns only (cross-lane/cross-wave merge afterwards).
  float runm[8], runs[8];
#pragma unroll
  for (int i = 0; i < 8; ++i) { runm[i] = NEG_BIG; runs[i] = 0.f; }

  for (int cb = wave; cb < LK_ / 16; cb += 8) {
    const int n0 = cb * 16;
    // K rows load identically to Q rows for the NT contraction.
    const float* krow = k + ((size_t)b * LK_ + n0 + lm) * D_;
    v16bf kh[2], kl[2];
    split_frag(krow + 0,  kbase, 1, kh[0], kl[0]);
    split_frag(krow + 32, kbase, 1, kh[1], kl[1]);

    v8f c;
#pragma unroll
    for (int i = 0; i < 8; ++i) c[i] = 0.f;
    c = wmma_bf16x3(c, qh[0], ql[0], kh[0], kl[0]);
    c = wmma_bf16x3(c, qh[1], ql[1], kh[1], kl[1]);

    // Epilogue on C layout (vgpr i -> row i + half*8, lane lm -> col).
#pragma unroll
    for (int i = 0; i < 8; ++i) {
      const int    m = i + half * 8;
      const size_t e = (bq0 + m) * LK_ + (size_t)(n0 + lm);
      float s = c[i] + f1[e] + f2[e] + f3[e] + f4[e] + f5[e];
      s *= INV_TEMP;
      if (mask[e]) s = NEG_BIG;
      attn[e] = s;  // raw score scratch; rewritten as probability in pass 2
      if (s > runm[i]) { runs[i] *= __expf(runm[i] - s); runm[i] = s; }
      runs[i] += __expf(s - runm[i]);
    }
  }

  // Merge (m,s) across the 16 lanes of each half (same rows, distinct cols).
#pragma unroll
  for (int off = 1; off < 16; off <<= 1) {
#pragma unroll
    for (int i = 0; i < 8; ++i) {
      float om = __shfl_xor(runm[i], off, 32);
      float os = __shfl_xor(runs[i], off, 32);
      float nm = fmaxf(runm[i], om);
      runs[i] = runs[i] * __expf(runm[i] - nm) + os * __expf(om - nm);
      runm[i] = nm;
    }
  }
  if (lm == 0) {
#pragma unroll
    for (int i = 0; i < 8; ++i) {
      red_m[wave][i + half * 8] = runm[i];
      red_s[wave][i + half * 8] = runs[i];
    }
  }
  __syncthreads();
  if (tid < 16) {
    float m = red_m[0][tid], s = red_s[0][tid];
#pragma unroll
    for (int w = 1; w < 8; ++w) {
      float om = red_m[w][tid], os = red_s[w][tid];
      float nm = fmaxf(m, om);
      s = s * __expf(m - nm) + os * __expf(om - nm);
      m = nm;
    }
    ((float2*)stats)[bq0 + tid] = make_float2(m, s);
  }
}

// ---------------------------------------------------------------------------
// Kernel 2: attn = softmax(S) written in place (L2-resident between passes),
//           O = attn @ V via bf16x3 WMMA. Grid/block as kernel 1;
//           wave w owns 32-wide k chunks w, w+8, ...; partial O reduced in LDS.
// ---------------------------------------------------------------------------
__global__ __launch_bounds__(256) void softmax_pv_kernel(
    const float* __restrict__ v, const float* __restrict__ stats,
    float* __restrict__ attn, float* __restrict__ out) {
  __shared__ float red[8 * 16 * 64];  // 32 KB

  const int b    = blockIdx.y;
  const int q0   = blockIdx.x * 16;
  const int tid  = threadIdx.x;
  const int wave = tid >> 5;
  const int lane = tid & 31;
  const int lm   = lane & 15;
  const int half = lane >> 4;
  const int kbase = half * 8;
  const size_t bq0 = (size_t)b * LQ_ + q0;

  const float2 st   = ((const float2*)stats)[bq0 + lm];
  const float  mrow = st.x;
  const float  invs = 1.0f / st.y;

  v8f acc[4];
#pragma unroll
  for (int nb = 0; nb < 4; ++nb)
#pragma unroll
    for (int i = 0; i < 8; ++i) acc[nb][i] = 0.f;

  float* prow = attn + (bq0 + lm) * LK_;

  for (int kc = wave; kc < LK_ / 32; kc += 8) {
    const int kk0 = kc * 32;

    // A-fragment of P: normalize score, write final attn, split to bf16.
    v16bf ph, pl;
#pragma unroll
    for (int j = 0; j < 16; ++j) {
      int   kk = kk0 + ((j >> 3) << 4) + kbase + (j & 7);
      float s  = prow[kk];
      float p  = __expf(s - mrow) * invs;
      prow[kk] = p;                    // each element written exactly once
      __bf16 h = (__bf16)p;
      ph[j] = h;
      pl[j] = (__bf16)(p - (float)h);
    }

    // B-fragments of V (32 x 16, four N blocks cover D=64); K walks rows.
    const float* vbase = v + ((size_t)b * LK_ + kk0) * D_;
#pragma unroll
    for (int nb = 0; nb < 4; ++nb) {
      v16bf vh, vl;
      split_frag(vbase + nb * 16 + lm, kbase, D_, vh, vl);
      acc[nb] = wmma_bf16x3(acc[nb], ph, pl, vh, vl);
    }
  }

  // Reduce 8 partial 16x64 tiles through LDS, then store O.
#pragma unroll
  for (int nb = 0; nb < 4; ++nb)
#pragma unroll
    for (int i = 0; i < 8; ++i)
      red[wave * 1024 + (i + half * 8) * 64 + nb * 16 + lm] = acc[nb][i];
  __syncthreads();
#pragma unroll
  for (int e = tid; e < 1024; e += 256) {
    float sum = 0.f;
#pragma unroll
    for (int w = 0; w < 8; ++w) sum += red[w * 1024 + e];
    out[(bq0 + (e >> 6)) * D_ + (e & 63)] = sum;
  }
}

// ---------------------------------------------------------------------------
extern "C" void kernel_launch(void* const* d_in, const int* in_sizes, int n_in,
                              void* d_out, int out_size, void* d_ws, size_t ws_size,
                              hipStream_t stream) {
  const float*   q    = (const float*)d_in[0];
  const float*   k    = (const float*)d_in[1];
  const float*   v    = (const float*)d_in[2];
  const uint8_t* mask = (const uint8_t*)d_in[3];  // jnp.bool_ -> 1 byte/elem
  const float*   f1   = (const float*)d_in[4];
  const float*   f2   = (const float*)d_in[5];
  const float*   f3   = (const float*)d_in[6];
  const float*   f4   = (const float*)d_in[7];
  const float*   f5   = (const float*)d_in[8];

  float* out  = (float*)d_out;                       // [B, LQ, D]
  float* attn = out + (size_t)B_ * LQ_ * D_;         // [B, LQ, LK]
  float* stats = (float*)d_ws;                       // [B*LQ] x {max, sum}

  dim3 grid(LQ_ / 16, B_);
  qk_scores_kernel<<<grid, 256, 0, stream>>>(q, k, mask, f1, f2, f3, f4, f5,
                                             attn, stats);
  softmax_pv_kernel<<<grid, 256, 0, stream>>>(v, stats, attn, out);
}
